// RNN_15779709845563
// MI455X (gfx1250) — matrix operand order repaired and checked by reading, not compiled, not run on hardware
//
#include <hip/hip_runtime.h>
#include <hip/hip_bf16.h>

// Problem constants (match reference)
#define TT  512
#define BB  64
#define IIN 128
#define HH  512
#define LL  3
#define OOUT 128

typedef __attribute__((ext_vector_type(16))) __bf16   v16bf;
typedef __attribute__((ext_vector_type(8)))  float    v8f;
typedef __attribute__((ext_vector_type(4)))  unsigned v4u;
typedef __attribute__((ext_vector_type(8)))  int      v8i;
typedef __attribute__((ext_vector_type(4)))  int      v4i;

static __device__ __forceinline__ __bf16 tobf(float f) { return (__bf16)f; }

// ---------------------------------------------------------------------------
// f32 -> bf16 conversion (stage x and all GEMM weights once)
// ---------------------------------------------------------------------------
__global__ void f32_to_bf16_kernel(const float* __restrict__ src,
                                   __bf16* __restrict__ dst, int n) {
    int i = blockIdx.x * blockDim.x + threadIdx.x;
    if (i < n) dst[i] = tobf(src[i]);
}

// ---------------------------------------------------------------------------
// WMMA GEMM: C[m,n] = sum_k A[m,k] * W[n,k] + bias[n]
//   A: bf16 [M,K] row-major, W: bf16 [N,K] row-major, C: f32 [M,N].
// Block = 256 threads = 8 waves.  Grid = (M/64, N/128).
// Wave w: N-tile = blockIdx.y*8 + w; computes 4 M-tiles (64 rows), reusing
// its B fragment 4x per K-chunk -> inner loop is 10x global_load_b128 +
// 4x v_wmma_f32_16x16x32_bf16 with 4 independent accumulator chains.
// A-fragment (ISA 7.12.2, 16-bit A 16x32): lane row = lane&15; elems 0..7 =
//   K khalf+0..7, elems 8..15 = K khalf+16..23, khalf = 8*(lane>=16).
// B-fragment (32x16): lane col = lane&15; same K order as A (khalf+0..7,
//   khalf+16..23) -> 2x b128.
// ---------------------------------------------------------------------------
__global__ __launch_bounds__(256) void gemm_bf16_wmma(
    const __bf16* __restrict__ A, const __bf16* __restrict__ W,
    const float* __restrict__ bias, float* __restrict__ C,
    int N, int K)
{
    const int wave = threadIdx.x >> 5;
    const int lane = threadIdx.x & 31;
    const int half = lane >> 4;
    const int n    = lane & 15;

    const int nt    = blockIdx.y * 8 + wave;
    const int m0row = blockIdx.x * 64;

    const __bf16* Ab = A + (size_t)(m0row + n) * K + half * 8;
    const __bf16* Wr = W + (size_t)(nt * 16 + n) * K + half * 8;

    v8f acc0 = {}, acc1 = {}, acc2 = {}, acc3 = {};
    for (int kc = 0; kc < (K >> 5); ++kc) {
        union { uint4 q[2]; v16bf v; } b, a0, a1, a2, a3;
        const uint4* pB = (const uint4*)(Wr + kc * 32);
        b.q[0] = pB[0]; b.q[1] = pB[2];
        const __bf16* Ak = Ab + kc * 32;
        const uint4* p0 = (const uint4*)(Ak);
        const uint4* p1 = (const uint4*)(Ak + (size_t)16 * K);
        const uint4* p2 = (const uint4*)(Ak + (size_t)32 * K);
        const uint4* p3 = (const uint4*)(Ak + (size_t)48 * K);
        a0.q[0] = p0[0]; a0.q[1] = p0[2];
        a1.q[0] = p1[0]; a1.q[1] = p1[2];
        a2.q[0] = p2[0]; a2.q[1] = p2[2];
        a3.q[0] = p3[0]; a3.q[1] = p3[2];
        acc0 = __builtin_amdgcn_wmma_f32_16x16x32_bf16(false, a0.v, false, b.v, (short)0, acc0, false, false);
        acc1 = __builtin_amdgcn_wmma_f32_16x16x32_bf16(false, a1.v, false, b.v, (short)0, acc1, false, false);
        acc2 = __builtin_amdgcn_wmma_f32_16x16x32_bf16(false, a2.v, false, b.v, (short)0, acc2, false, false);
        acc3 = __builtin_amdgcn_wmma_f32_16x16x32_bf16(false, a3.v, false, b.v, (short)0, acc3, false, false);
    }

    // C/D layout: lane col = lane&15; VGPR r -> row r + 8*half.
    const float bv = bias[nt * 16 + n];
    v8f accs[4] = { acc0, acc1, acc2, acc3 };
    #pragma unroll
    for (int m = 0; m < 4; ++m) {
        float* Cp = C + (size_t)(m0row + m * 16 + half * 8) * N + nt * 16 + n;
        #pragma unroll
        for (int r = 0; r < 8; ++r)
            Cp[(size_t)r * N] = accs[m][r] + bv;
    }
}

// ---------------------------------------------------------------------------
// TDM: async-load one contiguous 16x512 f32 row-block (32 KB) of `pre`
// into LDS.  Descriptor per ISA 8.3/8.4: group0 {count=1, lds_addr,
// global_addr, type=2}; group1 {data_size=4B, tensor_dim0=8192, dim1=1,
// tile_dim0=8192, tile_dim1=1, stride0=8192}.  Tracked by TENSORcnt.
// This toolchain's builtin takes 6 args (g0, g1, g2, g3, g4, cpol).
// ---------------------------------------------------------------------------
static __device__ __forceinline__ void tdm_load_pre(const float* gsrc,
                                                    unsigned ldsOff) {
    unsigned long long ga = (unsigned long long)(uintptr_t)gsrc;
    v4u g0 = { 1u,                                   // count=1 (valid)
               ldsOff,                               // lds_addr (bytes)
               (unsigned)ga,                         // global_addr[31:0]
               (unsigned)((ga >> 32) & 0x1FFFFFFu) | (2u << 30) }; // [56:32] | type=2
    v8i g1 = { (int)(2u << 16),      // data_size = 4B
               (int)(8192u << 16),   // tensor_dim0[15:0] in [31:16]
               (int)(1u << 16),      // tensor_dim0[31:16]=0, tensor_dim1[15:0]=1
               (int)(8192u << 16),   // tensor_dim1[31:16]=0, tile_dim0=8192
               (int)1,               // tile_dim1=1, tile_dim2=0
               (int)8192,            // tensor_dim0_stride[31:0]
               0, 0 };
    v4i z4 = { 0, 0, 0, 0 };
    v8i z8 = { 0, 0, 0, 0, 0, 0, 0, 0 };
    __builtin_amdgcn_tensor_load_to_lds(g0, g1, z4, z4, z8, 0);
}

// ---------------------------------------------------------------------------
// Recurrent scan for one layer.  Grid = 4 blocks (16 batch rows each; the
// recurrence is independent per batch row -> zero inter-block sync),
// block = 1024 threads = 32 waves.  Wave w owns output columns [16w,16w+16):
// its slice of W_hh^T (B-operand, K=512, 16 v16bf = 128 VGPRs) stays
// resident in registers for all 512 timesteps.  h lives in LDS (16 KB);
// pre[t] is double-buffered through LDS by the TDM, overlapped with WMMA.
//   h_new[b,:] = tanh(pre[t,b,:] + h[b,:] @ W_hh^T + b_hh)
// ---------------------------------------------------------------------------
__global__ __launch_bounds__(1024, 1) void rnn_scan_wmma(
    const float* __restrict__ pre,   // [T,B,H] (includes x@W_ih^T + b_ih)
    const float* __restrict__ Whh,   // [H,H] row-major (this layer)
    const float* __restrict__ bhh,   // [H]
    const float* __restrict__ h0,    // [B,H] (this layer)
    __bf16* __restrict__ ys,         // [T,B,H] bf16 layer output
    float*  __restrict__ hlast)      // [B,H] final hidden state
{
    __shared__ __bf16 hs[16 * HH];        // 16 KB: this block's h slice
    __shared__ float  preb[2][16 * HH];   // 2 x 32 KB: TDM double buffer

    const int tid  = threadIdx.x;
    const int wave = tid >> 5;       // 0..31
    const int lane = tid & 31;
    const int half = lane >> 4;
    const int n    = lane & 15;
    const int n0   = wave * 16;
    const int b0   = blockIdx.x * 16;
    const int mbase = half * 8;

    // Kick off TDM load of pre tile 0 while we set up.
    if (wave == 0)
        tdm_load_pre(pre + (size_t)b0 * HH,
                     (unsigned)(uintptr_t)&preb[0][0]);

    // init h slice in LDS
    for (int i = tid; i < 16 * HH; i += 1024)
        hs[i] = tobf(h0[(size_t)(b0 + (i >> 9)) * HH + (i & (HH - 1))]);

    // preload resident B fragments: B[k][n'] = Whh[(n0+n')*H + k]
    v16bf wb[16];
    {
        const float* Wr = Whh + (size_t)(n0 + n) * HH + half * 8;
        #pragma unroll
        for (int kc = 0; kc < 16; ++kc) {
            const float4* pW = (const float4*)(Wr + kc * 32);
            float4 w0 = pW[0], w1 = pW[1], w2 = pW[4], w3 = pW[5];
            v16bf b;
            b[0]=tobf(w0.x); b[1]=tobf(w0.y); b[2]=tobf(w0.z); b[3]=tobf(w0.w);
            b[4]=tobf(w1.x); b[5]=tobf(w1.y); b[6]=tobf(w1.z); b[7]=tobf(w1.w);
            b[8]=tobf(w2.x); b[9]=tobf(w2.y); b[10]=tobf(w2.z); b[11]=tobf(w2.w);
            b[12]=tobf(w3.x); b[13]=tobf(w3.y); b[14]=tobf(w3.z); b[15]=tobf(w3.w);
            wb[kc] = b;
        }
    }
    const float bval = bhh[n0 + n];
    const __bf16* arow = hs + (size_t)n * HH + half * 8;   // A row = lane&15

    __syncthreads();

    for (int t = 0; t < TT; ++t) {
        // Prefetch next step's pre tile via TDM (overlaps the WMMA chain).
        if (wave == 0 && t + 1 < TT)
            tdm_load_pre(pre + ((size_t)(t + 1) * BB + b0) * HH,
                         (unsigned)(uintptr_t)&preb[(t + 1) & 1][0]);

        v8f acc = {};
        #pragma unroll
        for (int kc = 0; kc < 16; ++kc) {
            union { uint4 q[2]; v16bf v; } a;
            const uint4* pA = (const uint4*)(arow + kc * 32);  // ds_load_b128 x2
            a.q[0] = pA[0];
            a.q[1] = pA[2];
            acc = __builtin_amdgcn_wmma_f32_16x16x32_bf16(
                false, a.v, false, wb[kc], (short)0, acc, false, false);
        }

        // Tile t done when TENSORcnt <= 1 (TDM completes in order; tile t+1
        // may remain in flight).  Barrier publishes it to all waves and also
        // closes this step's hs reads.  (wait argument must be a literal)
        if (wave == 0) {
            if (t + 1 < TT) {
                __builtin_amdgcn_s_wait_tensorcnt(1);
            } else {
                __builtin_amdgcn_s_wait_tensorcnt(0);
            }
        }
        __syncthreads();

        const float* prow = &preb[t & 1][(size_t)mbase * HH + (n0 + n)];
        float vals[8];
        #pragma unroll
        for (int r = 0; r < 8; ++r)
            vals[r] = tanhf(acc[r] + prow[(size_t)r * HH] + bval);

        #pragma unroll
        for (int r = 0; r < 8; ++r) {
            const int mrow = mbase + r;
            const __bf16 hv = tobf(vals[r]);
            hs[(size_t)mrow * HH + n0 + n] = hv;
            ys[(size_t)t * BB * HH + (size_t)(b0 + mrow) * HH + n0 + n] = hv;
        }
        if (t == TT - 1) {
            #pragma unroll
            for (int r = 0; r < 8; ++r)
                hlast[(size_t)(b0 + mbase + r) * HH + n0 + n] = vals[r];
        }
        __syncthreads();   // new h visible; also fences preb[t&1] reuse at t+1
    }
}

// ---------------------------------------------------------------------------
extern "C" void kernel_launch(void* const* d_in, const int* in_sizes, int n_in,
                              void* d_out, int out_size, void* d_ws, size_t ws_size,
                              hipStream_t stream) {
    const float* x     = (const float*)d_in[0];
    const float* h0    = (const float*)d_in[1];
    const float* W_ih0 = (const float*)d_in[2];
    const float* W_ihr = (const float*)d_in[3];
    const float* W_hh  = (const float*)d_in[4];
    const float* b_ih  = (const float*)d_in[5];
    const float* b_hh  = (const float*)d_in[6];
    const float* lin_W = (const float*)d_in[7];
    const float* lin_b = (const float*)d_in[8];

    float* out   = (float*)d_out;                    // [T,B,OUT]
    float* hlast = out + (size_t)TT * BB * OOUT;     // [L,B,H]

    char* ws = (char*)d_ws;
    __bf16* ysA = (__bf16*)ws;                                   // 32 MB
    __bf16* ysB = (__bf16*)(ws + (size_t)TT * BB * HH * 2);      // 32 MB
    float*  pre = (float*)(ws + (size_t)TT * BB * HH * 4);       // 64 MB
    __bf16* wts = (__bf16*)(ws + (size_t)TT * BB * HH * 8);      // bf16 weights
    __bf16* Wih0b = wts;                                // H*IN
    __bf16* Wih1b = Wih0b + (size_t)HH * IIN;           // H*H
    __bf16* Wih2b = Wih1b + (size_t)HH * HH;            // H*H
    __bf16* linWb = Wih2b + (size_t)HH * HH;            // OUT*H

    const int M = TT * BB;   // 32768

    // Stage x and all GEMM weights as bf16 (once).
    f32_to_bf16_kernel<<<(M * IIN + 255) / 256, 256, 0, stream>>>(x, ysA, M * IIN);
    f32_to_bf16_kernel<<<(HH * IIN + 255) / 256, 256, 0, stream>>>(W_ih0, Wih0b, HH * IIN);
    f32_to_bf16_kernel<<<(HH * HH + 255) / 256, 256, 0, stream>>>(W_ihr, Wih1b, HH * HH);
    f32_to_bf16_kernel<<<(HH * HH + 255) / 256, 256, 0, stream>>>(W_ihr + (size_t)HH * HH, Wih2b, HH * HH);
    f32_to_bf16_kernel<<<(OOUT * HH + 255) / 256, 256, 0, stream>>>(lin_W, linWb, OOUT * HH);

    const __bf16* Wihb[LL] = { Wih0b, Wih1b, Wih2b };

    __bf16* yin = ysA;
    __bf16* yout = ysB;
    for (int l = 0; l < LL; ++l) {
        const int K = (l == 0) ? IIN : HH;
        // pre = yin @ Wih^T + b_ih[l]   (M x H)
        gemm_bf16_wmma<<<dim3(M / 64, HH / 128), 256, 0, stream>>>(
            yin, Wihb[l], b_ih + (size_t)l * HH, pre, HH, K);
        // sequential scan (batch-parallel: 4 blocks x 16 rows)
        rnn_scan_wmma<<<4, 1024, 0, stream>>>(
            pre, W_hh + (size_t)l * HH * HH, b_hh + (size_t)l * HH,
            h0 + (size_t)l * BB * HH, yout, hlast + (size_t)l * BB * HH);
        __bf16* tmp = yin; yin = yout; yout = tmp;
    }

    // out = ys_last @ lin_W^T + lin_b   (M x OUT)
    gemm_bf16_wmma<<<dim3(M / 64, OOUT / 128), 256, 0, stream>>>(
        yin, linWb, lin_b, out, OOUT, HH);
}
